// Model_69432441307635
// MI455X (gfx1250) — compile-verified, matrix operand order checked by reading, not run on hardware
//
#include <hip/hip_runtime.h>
#include <hip/hip_bf16.h>
#include <math.h>

typedef __attribute__((ext_vector_type(16))) _Float16 v16h;
typedef __attribute__((ext_vector_type(8)))  float    v8f;
typedef __attribute__((ext_vector_type(4)))  int      v4i;

#define N_NODES 10000
#define NPAD    10112      // 79 * 128
#define EMB     256
#define DFEAT   32
#define NFAC    3
#define DMAP    64
#define NMOTIF  50000
#define EPSC    1e-5f
#define INV_TEMP 5.0f      // 1 / 0.2

#define BM 128
#define BN 128
#define BK 32
#define LDA 40             // padded LDS stride in halfs (80B, 16B aligned)
#define KSTEPS (EMB / BK)  // 8

// ---- CDNA5 async global->LDS path (guarded; falls back to sync copy) ----
#if defined(__has_builtin)
#  if __has_builtin(__builtin_amdgcn_global_load_async_to_lds_b128) && \
      __has_builtin(__builtin_amdgcn_s_wait_asynccnt)
#    define ASYNC_OK 1
#  endif
#endif
#ifndef ASYNC_OK
#  define ASYNC_OK 0
#endif

// b128 async copy: signature (per hipcc diagnostic) takes v4i pointers,
// global (AS1) source and LDS (AS3) destination, + imm offset / cpol.
typedef __attribute__((address_space(1))) v4i gv4i;
typedef __attribute__((address_space(3))) v4i lv4i;

__device__ __forceinline__ void async_cp16(const _Float16* g, _Float16* l) {
#if ASYNC_OK
    __builtin_amdgcn_global_load_async_to_lds_b128((gv4i*)g, (lv4i*)l, 0, 0);
#else
    *(float4*)l = *(const float4*)g;
#endif
}

// ---------------- products: normalize(feats[f], ks[f]) ----------------
__global__ void k_products(const float* __restrict__ feats, const float* __restrict__ ks,
                           float* __restrict__ prods) {
    int idx = blockIdx.x * blockDim.x + threadIdx.x;
    if (idx >= NFAC * N_NODES) return;
    int f = idx / N_NODES;
    int n = idx - f * N_NODES;
    const float* src = feats + (size_t)(f * N_NODES + n) * DFEAT;
    float v[DFEAT];
    float ss = 0.f;
    #pragma unroll
    for (int i = 0; i < DFEAT; ++i) { v[i] = src[i]; ss += v[i] * v[i]; }
    float radius = rsqrtf(fabsf(ks[f]));          // 1/sqrt(|k|)
    float scale  = 0.45f * radius * rsqrtf(ss);   // 0.9 * 0.5 * radius / ||x||
    float* dst = prods + (size_t)(f * N_NODES + n) * DFEAT;
    #pragma unroll
    for (int i = 0; i < DFEAT; ++i) dst[i] = v[i] * scale;
}

// ---------------- random mapping -> x2 [N, 256] ----------------
__global__ void k_x2(const float* __restrict__ prods, const float* __restrict__ feat_free,
                     const float* __restrict__ ks, const float* __restrict__ Ws,
                     const float* __restrict__ bias, const float* __restrict__ W_free,
                     const float* __restrict__ b_free, float* __restrict__ x2) {
    int gid = blockIdx.x * blockDim.x + threadIdx.x;
    if (gid >= N_NODES * EMB) return;
    int n = gid >> 8;
    int c = gid & 255;
    int f = c >> 6;
    int j = c & 63;
    float z;
    if (f < NFAC) {
        const float* xi = prods + (size_t)(f * N_NODES + n) * DFEAT;
        const float* w  = Ws + (size_t)(f * DMAP + j) * DFEAT;
        float div = 0.f, sx = 0.f;
        #pragma unroll
        for (int d = 0; d < DFEAT; ++d) {
            float xv = xi[d];
            float dd = xv - w[d];
            div += dd * dd;
            sx  += xv * xv;
        }
        float num  = 1.f + ks[f] * sx;
        float dist = __logf(num / (div + EPSC));
        z = __expf(15.5f * dist) * __cosf(dist + bias[f * DMAP + j]);
    } else {
        const float* xf = feat_free + (size_t)n * DFEAT;
        const float* w  = W_free + (size_t)j * DFEAT;
        float dot = 0.f;
        #pragma unroll
        for (int d = 0; d < DFEAT; ++d) dot += xf[d] * w[d];
        z = __expf(15.5f * dot) * __cosf(dot + b_free[j]);
    }
    x2[(size_t)n * EMB + c] = z;
}

// ---------------- row-normalize + f16 convert (padded rows -> 0) ----------------
__global__ void k_norm_f16(const float* __restrict__ src, _Float16* __restrict__ dst,
                           int nValid) {
    int row = blockIdx.x;
    int t = threadIdx.x;   // 256 == EMB
    __shared__ float red[256];
    float v = 0.f;
    if (row < nValid) v = src[(size_t)row * EMB + t];
    red[t] = v * v;
    __syncthreads();
    for (int s = 128; s > 0; s >>= 1) {
        if (t < s) red[t] += red[t + s];
        __syncthreads();
    }
    float rs = (row < nValid) ? rsqrtf(red[0] + 1e-30f) : 0.f;
    dst[(size_t)row * EMB + t] = (_Float16)(v * rs);
}

// ---------------- fused sim GEMM + exp + row/col reductions + diagonal ----------------
__global__ __launch_bounds__(256) void k_sim(const _Float16* __restrict__ xh,
                                             const _Float16* __restrict__ x2h,
                                             float* __restrict__ rowSums,
                                             float* __restrict__ colSums,
                                             float* __restrict__ pos) {
    __shared__ __align__(16) _Float16 As[2][BM * LDA];
    __shared__ __align__(16) _Float16 Bs[2][BN * LDA];
    int tid  = threadIdx.x;
    int lane = tid & 31;
    int wave = tid >> 5;
    int wr = wave >> 1;              // 0..3 : 32-row strip
    int wc = wave & 1;               // 0..1 : 64-col strip
    int rowBase = blockIdx.y * BM;
    int colBase = blockIdx.x * BN;

    v8f acc[2][4] = {};

    int ml = lane & 15;
    int kb = (lane & 16) ? 8 : 0;    // per-lane K chunk base (ISA 16-bit A/B layout)

    // async copy mapping: 128 rows x 32 halfs = 512 x 16B chunks per matrix,
    // 2 chunks per thread per matrix
    int r0 = tid >> 2;               // chunk 0 row (0..63)
    int o0 = (tid & 3) * 8;          // chunk 0 half-offset {0,8,16,24}
    int r1 = (tid + 256) >> 2;       // chunk 1 row (64..127)
    int o1 = o0;

    #define ISSUE_TILE(kt, buf)                                                          \
        do {                                                                             \
            int _k0 = (kt) * BK;                                                         \
            async_cp16(xh  + (size_t)(rowBase + r0) * EMB + _k0 + o0, &As[buf][r0 * LDA + o0]); \
            async_cp16(x2h + (size_t)(colBase + r0) * EMB + _k0 + o0, &Bs[buf][r0 * LDA + o0]); \
            async_cp16(xh  + (size_t)(rowBase + r1) * EMB + _k0 + o1, &As[buf][r1 * LDA + o1]); \
            async_cp16(x2h + (size_t)(colBase + r1) * EMB + _k0 + o1, &Bs[buf][r1 * LDA + o1]); \
        } while (0)

    ISSUE_TILE(0, 0);

    for (int kt = 0; kt < KSTEPS; ++kt) {
        int cur = kt & 1;
        if (kt + 1 < KSTEPS) ISSUE_TILE(kt + 1, cur ^ 1);
#if ASYNC_OK
        if (kt + 1 < KSTEPS) __builtin_amdgcn_s_wait_asynccnt(4);  // current tile landed
        else                 __builtin_amdgcn_s_wait_asynccnt(0);
#endif
        __syncthreads();   // all waves' copies of tile `cur` visible

        union Frag { v16h h; float4 q[2]; };
        Frag a[2], b[4];
        #pragma unroll
        for (int tm = 0; tm < 2; ++tm) {
            int m = wr * 32 + tm * 16 + ml;
            a[tm].q[0] = *(const float4*)&As[cur][m * LDA + kb];
            a[tm].q[1] = *(const float4*)&As[cur][m * LDA + kb + 16];
        }
        #pragma unroll
        for (int tn = 0; tn < 4; ++tn) {
            int n = wc * 64 + tn * 16 + ml;
            b[tn].q[0] = *(const float4*)&Bs[cur][n * LDA + kb];
            b[tn].q[1] = *(const float4*)&Bs[cur][n * LDA + kb + 16];
        }
        #pragma unroll
        for (int tm = 0; tm < 2; ++tm)
            #pragma unroll
            for (int tn = 0; tn < 4; ++tn)
                acc[tm][tn] = __builtin_amdgcn_wmma_f32_16x16x32_f16(
                    false, a[tm].h, false, b[tn].h, (short)0, acc[tm][tn], false, false);

        __syncthreads();   // done reading `cur` before it is overwritten
    }
    #undef ISSUE_TILE

    // epilogue: s = exp(sim/T); diagonal, row sums, column sums
    int hi = (lane >> 4) & 1;
    #pragma unroll
    for (int tm = 0; tm < 2; ++tm) {
        #pragma unroll
        for (int tn = 0; tn < 4; ++tn) {
            int gi0 = rowBase + wr * 32 + tm * 16;
            int gj  = colBase + wc * 64 + tn * 16 + ml;
            float csum = 0.f;
            #pragma unroll
            for (int v = 0; v < 8; ++v) {
                int i = gi0 + v + hi * 8;     // C layout: VGPR v -> row v (+8 for hi half)
                float s = 0.f;
                if (i < N_NODES && gj < N_NODES) {
                    s = __expf(acc[tm][tn][v] * INV_TEMP);
                    if (i == gj) pos[i] = s;
                }
                csum += s;
                float r = s;                  // reduce across the 16 lanes of this half
                r += __shfl_xor(r, 1, 32);
                r += __shfl_xor(r, 2, 32);
                r += __shfl_xor(r, 4, 32);
                r += __shfl_xor(r, 8, 32);
                if (ml == 0 && i < N_NODES) atomicAdd(&rowSums[i], r);
            }
            csum += __shfl_xor(csum, 16, 32); // combine the two 8-row halves per column
            if (hi == 0 && gj < N_NODES) atomicAdd(&colSums[gj], csum);
        }
    }
}

// ---------------- motif MLP + BCE ----------------
__global__ __launch_bounds__(256) void k_motif(const float* __restrict__ prods,
                                               const float* __restrict__ feat_free,
                                               const int* __restrict__ motif,
                                               const int* __restrict__ neg_uv,
                                               const float* __restrict__ W1,
                                               const float* __restrict__ b1,
                                               const float* __restrict__ W2,
                                               const float* __restrict__ b2,
                                               float* __restrict__ accOut) {
    __shared__ float W1s[96 * 64];
    __shared__ float b1s[64];
    __shared__ float W2s[64];
    __shared__ float red[256];
    int tid = threadIdx.x;
    for (int i = tid; i < 96 * 64; i += 256) W1s[i] = W1[i];
    if (tid < 64) { b1s[tid] = b1[tid]; W2s[tid] = W2[tid]; }
    __syncthreads();

    int pass  = blockIdx.y;      // 0..7 : 4 products x {pos,neg}
    int pf    = pass >> 1;
    int isNeg = pass & 1;
    int m = blockIdx.x * 256 + tid;
    float term = 0.f;
    if (m < NMOTIF) {
        int u, v, w;
        w = motif[2 * NMOTIF + m];
        if (isNeg) { u = neg_uv[m]; v = neg_uv[NMOTIF + m]; }
        else       { u = motif[m];  v = motif[NMOTIF + m]; }
        const float* P = (pf < NFAC) ? (prods + (size_t)pf * N_NODES * DFEAT) : feat_free;
        const float* pu = P + (size_t)u * DFEAT;
        const float* pv = P + (size_t)v * DFEAT;
        const float* pw = P + (size_t)w * DFEAT;
        float fv[96];
        #pragma unroll
        for (int i = 0; i < 32; ++i) { fv[i] = pu[i]; fv[32 + i] = pv[i]; fv[64 + i] = pw[i]; }
        float out = b2[0];
        for (int j = 0; j < 64; ++j) {
            float h = b1s[j];
            #pragma unroll
            for (int i = 0; i < 96; ++i) h += fv[i] * W1s[i * 64 + j];  // LDS broadcast
            h = fmaxf(h, 0.f);
            out += h * W2s[j];
        }
        float z  = isNeg ? -out : out;
        float ls = fminf(z, 0.f) - log1pf(__expf(-fabsf(z)));           // log_sigmoid
        term = -ls * (1.0f / NMOTIF);
    }
    red[tid] = term;
    __syncthreads();
    for (int s = 128; s > 0; s >>= 1) { if (tid < s) red[tid] += red[tid + s]; __syncthreads(); }
    if (tid == 0) atomicAdd(accOut, red[0]);
}

// ---------------- finalize scalar loss ----------------
__global__ void k_final(const float* __restrict__ rowSums, const float* __restrict__ colSums,
                        const float* __restrict__ pos, const float* __restrict__ motifAcc,
                        float* __restrict__ out) {
    __shared__ float red[256];
    int t = threadIdx.x;
    float acc = 0.f;
    for (int i = t; i < N_NODES; i += 256) {
        float p  = pos[i];
        float lp = __logf(p);
        acc += (__logf(colSums[i] - p) - lp);   // loss_1 term
        acc += (__logf(rowSums[i] - p) - lp);   // loss_2 term
    }
    red[t] = acc;
    __syncthreads();
    for (int s = 128; s > 0; s >>= 1) { if (t < s) red[t] += red[t + s]; __syncthreads(); }
    if (t == 0) out[0] = 0.5f * red[0] / (float)N_NODES + motifAcc[0];
}

extern "C" void kernel_launch(void* const* d_in, const int* in_sizes, int n_in,
                              void* d_out, int out_size, void* d_ws, size_t ws_size,
                              hipStream_t stream) {
    const float* x         = (const float*)d_in[0];
    const float* feats     = (const float*)d_in[1];
    const float* feat_free = (const float*)d_in[2];
    const float* ks        = (const float*)d_in[3];
    const float* Ws        = (const float*)d_in[4];
    const float* bias      = (const float*)d_in[5];
    const float* W_free    = (const float*)d_in[6];
    const float* b_free    = (const float*)d_in[7];
    const float* W1        = (const float*)d_in[8];
    const float* b1        = (const float*)d_in[9];
    const float* W2        = (const float*)d_in[10];
    const float* b2        = (const float*)d_in[11];
    const int*   motif     = (const int*)d_in[12];
    const int*   neg_uv    = (const int*)d_in[13];

    char* ws = (char*)d_ws;
    size_t off = 0;
    float* rowSums  = (float*)(ws + off); off += (size_t)N_NODES * sizeof(float);
    float* colSums  = (float*)(ws + off); off += (size_t)N_NODES * sizeof(float);
    float* pos      = (float*)(ws + off); off += (size_t)N_NODES * sizeof(float);
    float* motifAcc = (float*)(ws + off); off += 256;
    off = (off + 255) & ~(size_t)255;
    float* prods = (float*)(ws + off); off += (size_t)NFAC * N_NODES * DFEAT * sizeof(float);
    off = (off + 255) & ~(size_t)255;
    float* x2 = (float*)(ws + off); off += (size_t)N_NODES * EMB * sizeof(float);
    off = (off + 255) & ~(size_t)255;
    _Float16* xh  = (_Float16*)(ws + off); off += (size_t)NPAD * EMB * sizeof(_Float16);
    _Float16* x2h = (_Float16*)(ws + off); off += (size_t)NPAD * EMB * sizeof(_Float16);

    // zero accumulators (rowSums, colSums, pos, motifAcc region)
    (void)hipMemsetAsync(ws, 0, (size_t)3 * N_NODES * sizeof(float) + 256, stream);

    k_products<<<(NFAC * N_NODES + 255) / 256, 256, 0, stream>>>(feats, ks, prods);
    k_x2<<<(N_NODES * EMB + 255) / 256, 256, 0, stream>>>(prods, feat_free, ks, Ws, bias,
                                                          W_free, b_free, x2);
    k_norm_f16<<<NPAD, 256, 0, stream>>>(x,  xh,  N_NODES);
    k_norm_f16<<<NPAD, 256, 0, stream>>>(x2, x2h, N_NODES);

    dim3 gSim((N_NODES + BN - 1) / BN, (N_NODES + BM - 1) / BM);  // 79 x 79
    k_sim<<<gSim, 256, 0, stream>>>(xh, x2h, rowSums, colSums, pos);

    dim3 gMot((NMOTIF + 255) / 256, 8);
    k_motif<<<gMot, 256, 0, stream>>>(prods, feat_free, motif, neg_uv,
                                      W1, b1, W2, b2, motifAcc);

    k_final<<<1, 256, 0, stream>>>(rowSums, colSums, pos, motifAcc, (float*)d_out);
}
